// BitDanceFP8ScaledLinear_20607253086856
// MI455X (gfx1250) — matrix operand order repaired and checked by reading, not compiled, not run on hardware
//
#include <hip/hip_runtime.h>
#include <hip/hip_bf16.h>

typedef __attribute__((ext_vector_type(16))) __bf16 v16bf;
typedef __attribute__((ext_vector_type(8)))  float  v8f;

#define BM   128
#define BN   128
#define BK   32
#define LDSK (BK + 8)          // padded LDS stride in bf16 elements (80 B, 16B-aligned)

// fp32 pair -> packed 2x bf16 in one/three VALU ops
__device__ __forceinline__ unsigned pack2_bf16(float a, float b) {
#if __has_builtin(__builtin_amdgcn_cvt_pk_bf16_f32)
    typedef __attribute__((ext_vector_type(2))) __bf16 v2bf;
    v2bf r = __builtin_amdgcn_cvt_pk_bf16_f32(a, b);
    return __builtin_bit_cast(unsigned, r);
#else
    // round-bias then splice the two high halves with a single v_perm_b32
    unsigned ua = __float_as_uint(a) + 0x8000u;
    unsigned ub = __float_as_uint(b) + 0x8000u;
    return __builtin_amdgcn_perm(ub, ua, 0x07060302u);
#endif
}

__global__ __launch_bounds__(256)
void fp8_scaled_linear_wmma(const float* __restrict__ X,   // [M, K]
                            const float* __restrict__ W,   // [N, K]
                            const float* __restrict__ Sc,  // [N]
                            const float* __restrict__ Bi,  // [N]
                            float* __restrict__ Y,         // [M, N]
                            int M, int N, int K)
{
    __shared__ __align__(16) __bf16 As[2][BM * LDSK];
    __shared__ __align__(16) __bf16 Bs[2][BN * LDSK];

    const int tid  = threadIdx.x;
    const int lane = tid & 31;
    const int wave = tid >> 5;       // 0..7
    const int wm   = wave >> 2;      // 0..1  (wave tile: 64 M x 32 N)
    const int wn   = wave & 3;       // 0..3

    const int n0 = blockIdx.x * BN;
    const int m0 = blockIdx.y * BM;

    // ---- global staging: 8 consecutive floats per thread per row-chunk ----
    const int lrow = tid >> 2;        // 0..63
    const int lcol = (tid & 3) * 8;   // 0,8,16,24
    const float* xg = X + (size_t)(m0 + lrow) * K + lcol;
    const float* wg = W + (size_t)(n0 + lrow) * K + lcol;

    float4 xr[4], wr[4];              // [chunk(2)][half(2)]
    const int nk = K / BK;

    auto do_load = [&](int kt) {
        const size_t ko = (size_t)kt * BK;
        #pragma unroll
        for (int i = 0; i < 2; ++i) {
            const float* px = xg + (size_t)(i * 64) * K + ko;
            const float* pw = wg + (size_t)(i * 64) * K + ko;
            xr[2*i]   = *(const float4*)(px);
            xr[2*i+1] = *(const float4*)(px + 4);
            wr[2*i]   = *(const float4*)(pw);
            wr[2*i+1] = *(const float4*)(pw + 4);
        }
    };
    auto do_store = [&](int buf) {
        #pragma unroll
        for (int i = 0; i < 2; ++i) {
            uint4 va;
            va.x = pack2_bf16(xr[2*i].x,   xr[2*i].y);
            va.y = pack2_bf16(xr[2*i].z,   xr[2*i].w);
            va.z = pack2_bf16(xr[2*i+1].x, xr[2*i+1].y);
            va.w = pack2_bf16(xr[2*i+1].z, xr[2*i+1].w);
            *(uint4*)&As[buf][(lrow + i * 64) * LDSK + lcol] = va;  // ds_store_b128
            uint4 vb;
            vb.x = pack2_bf16(wr[2*i].x,   wr[2*i].y);
            vb.y = pack2_bf16(wr[2*i].z,   wr[2*i].w);
            vb.z = pack2_bf16(wr[2*i+1].x, wr[2*i+1].y);
            vb.w = pack2_bf16(wr[2*i+1].z, wr[2*i+1].w);
            *(uint4*)&Bs[buf][(lrow + i * 64) * LDSK + lcol] = vb;  // ds_store_b128
        }
    };

    // ---- fragment loaders (ISA VGPR layouts, two ds_load_b128 each) ----
    const int frow = lane & 15;
    const int hi   = lane >> 4;       // lane group 0/1

    // A 16x32: lanes 0-15 K{0-7,16-23}, lanes 16-31 K{8-15,24-31}
    auto load_a = [&](const __bf16* base, int row) -> v16bf {
        const __bf16* p = base + (size_t)row * LDSK + hi * 8;
        union { uint4 u[2]; v16bf v; } t;
        t.u[0] = *(const uint4*)(p);
        t.u[1] = *(const uint4*)(p + 16);
        return t.v;
    };
    // B 32x16: lanes 0-15 K{0-7,16-23}, lanes 16-31 K{8-15,24-31}
    auto load_b = [&](const __bf16* base, int row) -> v16bf {
        const __bf16* p = base + (size_t)row * LDSK + hi * 8;
        union { uint4 u[2]; v16bf v; } t;
        t.u[0] = *(const uint4*)(p);
        t.u[1] = *(const uint4*)(p + 16);
        return t.v;
    };

    v8f acc[4][2] = {};

    do_load(0);
    do_store(0);
    __syncthreads();

    for (int kt = 0; kt < nk; ++kt) {
        const int cur = kt & 1;

        if (kt + 1 < nk) do_load(kt + 1);
        if (kt + 2 < nk) {
            const size_t ko = (size_t)(kt + 2) * BK;
            __builtin_prefetch(xg + ko, 0, 1);   // global_prefetch_b8
            __builtin_prefetch(wg + ko, 0, 1);
        }

        v16bf afrag[4], bfrag[2];
        #pragma unroll
        for (int mt = 0; mt < 4; ++mt)
            afrag[mt] = load_a(&As[cur][0], wm * 64 + mt * 16 + frow);
        #pragma unroll
        for (int nt = 0; nt < 2; ++nt)
            bfrag[nt] = load_b(&Bs[cur][0], wn * 32 + nt * 16 + frow);

        #pragma unroll
        for (int mt = 0; mt < 4; ++mt)
            #pragma unroll
            for (int nt = 0; nt < 2; ++nt)
                acc[mt][nt] = __builtin_amdgcn_wmma_f32_16x16x32_bf16(
                    false, afrag[mt], false, bfrag[nt],
                    (short)0, acc[mt][nt], false, false);

        if (kt + 1 < nk) do_store(cur ^ 1);
        __syncthreads();
    }

    // ---- epilogue: fold per-channel scale + bias, store fp32 ----
    #pragma unroll
    for (int nt = 0; nt < 2; ++nt) {
        const int n = n0 + wn * 32 + nt * 16 + frow;
        const float s = Sc[n];
        const float b = Bi[n];
        #pragma unroll
        for (int mt = 0; mt < 4; ++mt) {
            const int mbase = m0 + wm * 64 + mt * 16 + hi * 8;   // VGPR i -> row mbase+i
            float* yp = Y + (size_t)mbase * N + n;
            #pragma unroll
            for (int i = 0; i < 8; ++i)
                yp[(size_t)i * N] = acc[mt][nt][i] * s + b;
        }
    }
}

extern "C" void kernel_launch(void* const* d_in, const int* in_sizes, int n_in,
                              void* d_out, int out_size, void* d_ws, size_t ws_size,
                              hipStream_t stream) {
    const float* x     = (const float*)d_in[0];   // [B,S,IN] fp32
    const float* w     = (const float*)d_in[1];   // [OUT,IN] fp32
    const float* scale = (const float*)d_in[2];   // [OUT]    fp32
    const float* bias  = (const float*)d_in[3];   // [OUT]    fp32
    float* y = (float*)d_out;                     // [B,S,OUT] fp32

    const int OUT = in_sizes[2];
    const int IN  = in_sizes[1] / OUT;
    const int M   = in_sizes[0] / IN;             // B*S

    dim3 grid(OUT / BN, M / BM);                  // 86 x 64 for the reference shape
    fp8_scaled_linear_wmma<<<grid, 256, 0, stream>>>(x, w, scale, bias, y, M, OUT, IN);
}
